// CTCLoss_84052509982854
// MI455X (gfx1250) — compile-verified, matrix operand order checked by reading, not run on hardware
//
#include <hip/hip_runtime.h>
#include <hip/hip_bf16.h>

typedef __attribute__((ext_vector_type(16))) _Float16 v16h;
typedef __attribute__((ext_vector_type(8)))  float    v8f;

#define TT   512
#define NN   64
#define LL   1024
#define SS   50
#define EE   101           // 2*S + 1
#define KTOT (NN * SS)
#define ROWS (TT * NN)
#define NEGV (-1.0e30f)

// ---------------------------------------------------------------------------
// Kernel 1: lse[t*N+n] = max_l acts[t,n,l] + log(sum_l exp(acts[t,n,l]-max))
// 16 rows per wave; Sigma-exp done by v_wmma_f32_16x16x32_f16 against a ones
// B-matrix (so C[m][*] = row-sum of A, independent of K-slot placement).
// ---------------------------------------------------------------------------
__device__ __forceinline__ float pick8(v8f c, int j) {
    float v = c[0];
    v = (j == 1) ? c[1] : v;
    v = (j == 2) ? c[2] : v;
    v = (j == 3) ? c[3] : v;
    v = (j == 4) ? c[4] : v;
    v = (j == 5) ? c[5] : v;
    v = (j == 6) ? c[6] : v;
    v = (j == 7) ? c[7] : v;
    return v;
}

__global__ __launch_bounds__(256) void ctc_lse_kernel(
    const float* __restrict__ acts, float* __restrict__ lse)
{
    const int wave   = threadIdx.x >> 5;
    const int lane   = threadIdx.x & 31;
    const int rgroup = blockIdx.x * 8 + wave;     // 16-row group
    const int rowBase = rgroup * 16;
    const int r    = lane & 15;                   // my row within group
    const int half = lane >> 4;                   // 0: low cols, 1: high cols
    const float* __restrict__ rowp = acts + (size_t)(rowBase + r) * LL;

    // ---- pass 1: per-row max (each lane scans half the row) ----
    float m = -3.0e38f;
    const float* p = rowp + half * 512;
    #pragma unroll 4
    for (int i = 0; i < 512; i += 4) {
        float4 v = *(const float4*)(p + i);
        m = fmaxf(m, fmaxf(fmaxf(v.x, v.y), fmaxf(v.z, v.w)));
    }
    m = fmaxf(m, __shfl_xor(m, 16, 32));          // combine the two halves

    // ---- pass 2: exp + WMMA row-sum accumulate (f32 accumulator) ----
    v16h ones;
    #pragma unroll
    for (int i = 0; i < 16; ++i) ones[i] = (_Float16)1.0f;

    v8f acc = {};
    for (int c = 0; c < LL; c += 32) {
        const float* q = rowp + c + half * 16;    // 16 contiguous floats/lane
        v16h a;
        #pragma unroll
        for (int i = 0; i < 16; i += 4) {
            float4 v = *(const float4*)(q + i);
            a[i + 0] = (_Float16)__expf(v.x - m);
            a[i + 1] = (_Float16)__expf(v.y - m);
            a[i + 2] = (_Float16)__expf(v.z - m);
            a[i + 3] = (_Float16)__expf(v.w - m);
        }
        acc = __builtin_amdgcn_wmma_f32_16x16x32_f16(
            /*neg_a=*/false, a, /*neg_b=*/false, ones,
            /*c_mod=*/(short)0, acc, /*reuse_a=*/false, /*reuse_b=*/false);
    }

    // ---- extract row sums from C layout ----
    // VGPR j: lanes 0-15 -> M=j, lanes 16-31 -> M=8+j
    int myrow = (lane < 8) ? lane
              : ((lane >= 16 && lane < 24) ? (lane - 8) : (lane & 15));
    float mr  = __shfl(m, myrow & 15, 32);        // full-EXEC shuffle
    float sum = pick8(acc, lane & 7);
    if (lane < 8)
        lse[rowBase + lane] = mr + __logf(sum);
    else if (lane >= 16 && lane < 24)
        lse[rowBase + lane - 8] = mr + __logf(sum);
}

// ---------------------------------------------------------------------------
// Kernel 2: per-sample CTC forward recursion. One block per sample n,
// one thread per extended state, LDS ping-pong alpha, prefetched gathers.
// ---------------------------------------------------------------------------
__global__ __launch_bounds__(128) void ctc_alpha_kernel(
    const float* __restrict__ acts, const int* __restrict__ labels,
    const int* __restrict__ acts_lens, const int* __restrict__ labels_lens,
    const float* __restrict__ lse, float* __restrict__ costs)
{
    const int n   = blockIdx.x;
    const int tid = threadIdx.x;

    __shared__ int   s_ext[EE];
    __shared__ unsigned char s_cs[EE];
    __shared__ float bufA[EE], bufB[EE];
    __shared__ int   s_off, s_len, s_alen;

    if (tid == 0) {
        int o = 0;
        for (int i = 0; i < n; ++i) o += labels_lens[i];  // exclusive cumsum
        s_off = o; s_len = labels_lens[n]; s_alen = acts_lens[n];
    }
    __syncthreads();
    const int off = s_off, len = s_len, alen = s_alen;

    // extended label sequence: blank, l1, blank, ..., lS, blank
    if (tid < EE) {
        int lab = 0;
        if (tid & 1) {
            int s = tid >> 1;
            if (s < len) {
                int idx = off + s;
                if (idx > KTOT - 1) idx = KTOT - 1;
                lab = labels[idx];
            }
        }
        s_ext[tid] = lab;
    }
    __syncthreads();
    if (tid < EE) {
        int lab = s_ext[tid];
        s_cs[tid] = (unsigned char)((lab != 0) && (tid < 2 || lab != s_ext[tid - 2]));
    }
    __syncthreads();

    const int  myext = (tid < EE) ? s_ext[tid] : 0;
    const bool mycs  = (tid < EE) ? (s_cs[tid] != 0) : false;
    const size_t frame = (size_t)NN * LL;

    // t = 0 init
    if (tid < EE) {
        float lp0 = acts[(size_t)n * LL + myext] - lse[n];
        bufA[tid] = (tid < 2) ? lp0 : NEGV;
    }
    __syncthreads();

    // prefetch frame t=1
    float act_n = 0.f;
    if (tid < EE) act_n = acts[frame + (size_t)n * LL + myext];
    float lse_n = lse[NN + n];

    float* prev = bufA;
    float* cur  = bufB;
    for (int t = 1; t < TT; ++t) {
        float lpc = act_n - lse_n;
        if (t + 1 < TT) {                      // prefetch next frame's gathers
            size_t base = (size_t)(t + 1) * frame + (size_t)n * LL;
            if (tid < EE) act_n = acts[base + myext];
            lse_n = lse[(t + 1) * NN + n];
        }
        if (tid < EE) {
            float a0 = prev[tid];
            float a1 = (tid >= 1) ? prev[tid - 1] : NEGV;
            float a2 = (tid >= 2 && mycs) ? prev[tid - 2] : NEGV;
            float mx = fmaxf(a0, fmaxf(a1, a2));
            float nv = mx + __logf(__expf(a0 - mx) + __expf(a1 - mx) + __expf(a2 - mx)) + lpc;
            cur[tid] = (t < alen) ? nv : a0;   // freeze past sequence length
        }
        __syncthreads();
        float* tmp = prev; prev = cur; cur = tmp;
    }

    if (tid == 0) {
        int   eend = 2 * len;
        float ae = prev[eend];
        float al = prev[eend - 1];
        float mm = fmaxf(ae, al);
        float ll = mm + __logf(__expf(ae - mm) + __expf(al - mm));
        costs[n] = -ll;
    }
}

// ---------------------------------------------------------------------------
// Kernel 3: deterministic fixed-order sum of the 64 costs -> scalar output.
// ---------------------------------------------------------------------------
__global__ void ctc_sum_kernel(const float* __restrict__ costs, float* __restrict__ out)
{
    if (threadIdx.x == 0 && blockIdx.x == 0) {
        float s = 0.f;
        for (int i = 0; i < NN; ++i) s += costs[i];
        out[0] = s;
    }
}

extern "C" void kernel_launch(void* const* d_in, const int* in_sizes, int n_in,
                              void* d_out, int out_size, void* d_ws, size_t ws_size,
                              hipStream_t stream) {
    const float* acts        = (const float*)d_in[0];
    const int*   labels      = (const int*)d_in[1];
    const int*   acts_lens   = (const int*)d_in[2];
    const int*   labels_lens = (const int*)d_in[3];

    float* lse   = (float*)d_ws;          // ROWS floats (128 KB)
    float* costs = lse + ROWS;            // NN floats

    ctc_lse_kernel<<<ROWS / 128, 256, 0, stream>>>(acts, lse);
    ctc_alpha_kernel<<<NN, 128, 0, stream>>>(acts, labels, acts_lens, labels_lens, lse, costs);
    ctc_sum_kernel<<<1, 32, 0, stream>>>(costs, (float*)d_out);
}
